// MambaLayer_image_20392504722025
// MI455X (gfx1250) — matrix operand order repaired and checked by reading, not compiled, not run on hardware
//
#include <hip/hip_runtime.h>
#include <math.h>

// ---------------- problem constants (from reference) ----------------
#define LSEQ   13824      // 24*24*24
#define CCH    128        // C
#define DMH    64         // DM
#define DIH    128        // DI
#define DSN    8          // DS
#define DTRN   4          // DTR
#define CHLEN  64         // scan chunk length
#define NCHUNK 216        // LSEQ / CHLEN

typedef __attribute__((ext_vector_type(16))) _Float16 v16h;
typedef __attribute__((ext_vector_type(8)))  _Float16 v8h;
typedef __attribute__((ext_vector_type(8)))  float    v8f;

// map canonical sequence index l -> spatial offset in (D,H,W) flat order for a mode
__device__ __forceinline__ int spatial_off(int mode, int l) {
  int i2 = l % 24; int t = l / 24; int i1 = t % 24; int i0 = t / 24;
  int d, h, w;
  if (mode == 0)      { d = i0; h = i1; w = i2; }
  else if (mode == 1) { h = i0; w = i1; d = i2; }
  else                { w = i0; d = i1; h = i2; }
  return (d * 24 + h) * 24 + w;
}

__device__ __forceinline__ float silu_f(float v) { return v / (1.f + expf(-v)); }
__device__ __forceinline__ float softplus_f(float v) {
  return (v > 20.f) ? v : log1pf(expf(v));
}

// ---------------- f32 -> f16 conversion (weights) ----------------
__global__ void k_cvt16(const float* __restrict__ src, _Float16* __restrict__ dst, int n) {
  int i = blockIdx.x * blockDim.x + threadIdx.x;
  if (i < n) dst[i] = (_Float16)src[i];
}

// ---------------- layernorm + split + reverse-prep (writes f16 halves) ----------------
__global__ void k_prep_ln(const float* __restrict__ cur,
                          _Float16* __restrict__ xa, _Float16* __restrict__ xbr,
                          const float* __restrict__ lng, const float* __restrict__ lnb,
                          int mode) {
  __shared__ float red[CCH];
  int l = blockIdx.x;
  int c = threadIdx.x;
  int off = spatial_off(mode, l);
  float v = cur[c * LSEQ + off];
  red[c] = v; __syncthreads();
  for (int o = 64; o > 0; o >>= 1) { if (c < o) red[c] += red[c + o]; __syncthreads(); }
  float mean = red[0] * (1.f / CCH);
  __syncthreads();
  float dv = (v - mean) * (v - mean);
  red[c] = dv; __syncthreads();
  for (int o = 64; o > 0; o >>= 1) { if (c < o) red[c] += red[c + o]; __syncthreads(); }
  float var = red[0] * (1.f / CCH);
  float xn = (v - mean) * rsqrtf(var + 1e-5f) * lng[c] + lnb[c];
  if (c < DMH) xa[l * DMH + c] = (_Float16)xn;
  else         xbr[(LSEQ - 1 - l) * DMH + (c - DMH)] = (_Float16)xn;
}

// ---------------- WMMA GEMM: C(MxN,f32) = A(MxK,f16) * W(NxK,f16)^T ----------------
// 4 waves / block; wave owns one 16-row M-tile and NT 16-col N-tiles.
// Fully unrolled (K compile-time) -> dense v_wmma block, A fragment reused NT times.
template<int K, int NT>
__global__ void k_gemm_h(const _Float16* __restrict__ A,
                         const _Float16* __restrict__ W,
                         float* __restrict__ Cmat, int ldc, int M, int rev) {
  int lane = threadIdx.x & 31;
  int wv   = threadIdx.x >> 5;
  int m0   = (blockIdx.x * 4 + wv) * 16;
  int n0   = blockIdx.y * (NT * 16);
  int g  = lane >> 4;          // half-wave group
  int ln = lane & 15;
  // A 16x32 f16 layout: lanes0-15 hold K {0..7,16..23}; lanes16-31 hold {8..15,24..31}
  int c0 = g ? 8 : 0;
  int c1 = g ? 24 : 16;
  // B 32x16 f16 layout: same K order as A; lanes0-15 hold K {0..7,16..23}; lanes16-31 hold {8..15,24..31}
  int bo = g ? 8 : 0;
  const _Float16* ap = A + (m0 + ln) * K;
  const _Float16* wp = W + (n0 + ln) * K + bo;
  v8f zero = {};
  v8f acc[NT];
#pragma unroll
  for (int t = 0; t < NT; ++t) acc[t] = zero;
#pragma unroll
  for (int k0 = 0; k0 < K; k0 += 32) {
    __builtin_prefetch(ap + k0 + 32, 0, 1);     // -> global_prefetch_b8
    v8h alo = *(const v8h*)(ap + k0 + c0);
    v8h ahi = *(const v8h*)(ap + k0 + c1);
    v16h av = __builtin_shufflevector(alo, ahi, 0,1,2,3,4,5,6,7,8,9,10,11,12,13,14,15);
#pragma unroll
    for (int t = 0; t < NT; ++t) {
      const _Float16* wq = wp + t * (16 * K) + k0;
      v8h blo = *(const v8h*)(wq);
      v8h bhi = *(const v8h*)(wq + 16);
      v16h bv = __builtin_shufflevector(blo, bhi, 0,1,2,3,4,5,6,7,8,9,10,11,12,13,14,15);
      acc[t] = __builtin_amdgcn_wmma_f32_16x16x32_f16(false, av, false, bv,
                                                      (short)0, acc[t], false, false);
    }
  }
#pragma unroll
  for (int t = 0; t < NT; ++t) {
#pragma unroll
    for (int r = 0; r < 8; ++r) {
      int row  = m0 + r + (g ? 8 : 0);
      int orow = rev ? (M - 1 - row) : row;
      Cmat[orow * ldc + n0 + t * 16 + ln] = acc[t][r];
    }
  }
}

// ---------------- causal depthwise conv (k=4) + bias + SiLU ----------------
__global__ void k_conv_silu(const float* __restrict__ xz,
                            const float* __restrict__ cw, const float* __restrict__ cb,
                            float* __restrict__ xc) {
  int idx = blockIdx.x * blockDim.x + threadIdx.x;   // over LSEQ*DIH
  int l = idx >> 7, d = idx & 127;
  float acc = cb[d];
#pragma unroll
  for (int t = 0; t < 4; ++t) {
    int ls = l - 3 + t;
    if (ls >= 0) acc += xz[ls * 256 + d] * cw[d * 4 + t];
  }
  xc[l * DIH + d] = silu_f(acc);
}

// ---------------- x-proj (N=20) + dt-proj + softplus, one block per l ----------------
__global__ void k_proj_dt(const float* __restrict__ xc,
                          const float* __restrict__ xpw,
                          const float* __restrict__ dpw, const float* __restrict__ dpb,
                          float* __restrict__ dbc, float* __restrict__ dt) {
  __shared__ float row[DIH];
  __shared__ float dbs[20];
  int l = blockIdx.x, t = threadIdx.x;
  row[t] = xc[l * DIH + t];
  __syncthreads();
  if (t < 20) {
    float acc = 0.f;
    const float* wr = xpw + t * DIH;
    for (int d = 0; d < DIH; ++d) acc += row[d] * wr[d];
    dbs[t] = acc;
    dbc[l * 20 + t] = acc;
  }
  __syncthreads();
  float acc = dpb[t];
#pragma unroll
  for (int r = 0; r < DTRN; ++r) acc += dbs[r] * dpw[t * DTRN + r];
  dt[l * DIH + t] = softplus_f(acc);
}

// ---------------- scan pass 1: per-chunk affine aggregates ----------------
__global__ void k_scan1(const float* __restrict__ dt, const float* __restrict__ xc,
                        const float* __restrict__ dbc, const float* __restrict__ alog,
                        float* __restrict__ aggA, float* __restrict__ aggB) {
  __shared__ float sB[CHLEN * DSN];
  int ch = blockIdx.x, d = threadIdx.x;
  int base = ch * CHLEN;
  for (int i = d; i < CHLEN * DSN; i += DIH) {
    int li = i >> 3, s = i & 7;
    sB[i] = dbc[(base + li) * 20 + DTRN + s];
  }
  __syncthreads();
  float Av[DSN], aP[DSN], bS[DSN];
#pragma unroll
  for (int s = 0; s < DSN; ++s) { Av[s] = -expf(alog[d * DSN + s]); aP[s] = 1.f; bS[s] = 0.f; }
  for (int li = 0; li < CHLEN; ++li) {
    int Lx = base + li;
    float dtv = dt[Lx * DIH + d];
    float xcv = xc[Lx * DIH + d];
#pragma unroll
    for (int s = 0; s < DSN; ++s) {
      float a = expf(dtv * Av[s]);
      float b = dtv * sB[li * DSN + s] * xcv;
      bS[s] = a * bS[s] + b;
      aP[s] *= a;
    }
  }
  int o = ch * 1024 + d * DSN;
#pragma unroll
  for (int s = 0; s < DSN; ++s) { aggA[o + s] = aP[s]; aggB[o + s] = bS[s]; }
}

// ---------------- scan pass 2: combine chunk aggregates -> exclusive chunk prefix ----------------
__global__ void k_scan2(const float* __restrict__ aggA, float* __restrict__ aggB) {
  int t = blockIdx.x * blockDim.x + threadIdx.x;   // 0..1023 = d*8+s
  float h = 0.f;
  for (int c = 0; c < NCHUNK; ++c) {
    int idx = c * 1024 + t;
    float a = aggA[idx];
    float b = aggB[idx];
    aggB[idx] = h;
    h = a * h + b;
  }
}

// ---------------- scan pass 3: re-scan, y = sum_s h*C + D*xc, gate silu(z), write f16 ----------------
__global__ void k_scan3(const float* __restrict__ dt, const float* __restrict__ xc,
                        const float* __restrict__ dbc, const float* __restrict__ alog,
                        const float* __restrict__ dpar, const float* __restrict__ pref,
                        const float* __restrict__ xz, _Float16* __restrict__ yg) {
  __shared__ float sB[CHLEN * DSN];
  __shared__ float sC[CHLEN * DSN];
  int ch = blockIdx.x, d = threadIdx.x;
  int base = ch * CHLEN;
  for (int i = d; i < CHLEN * DSN; i += DIH) {
    int li = i >> 3, s = i & 7;
    sB[i] = dbc[(base + li) * 20 + DTRN + s];
    sC[i] = dbc[(base + li) * 20 + DTRN + DSN + s];
  }
  __syncthreads();
  float Av[DSN], h[DSN];
  int po = ch * 1024 + d * DSN;
#pragma unroll
  for (int s = 0; s < DSN; ++s) { Av[s] = -expf(alog[d * DSN + s]); h[s] = pref[po + s]; }
  float Dp = dpar[d];
  for (int li = 0; li < CHLEN; ++li) {
    int Lx = base + li;
    float dtv = dt[Lx * DIH + d];
    float xcv = xc[Lx * DIH + d];
    float y = Dp * xcv;
#pragma unroll
    for (int s = 0; s < DSN; ++s) {
      float a = expf(dtv * Av[s]);
      float b = dtv * sB[li * DSN + s] * xcv;
      h[s] = a * h[s] + b;
      y += h[s] * sC[li * DSN + s];
    }
    float z = xz[Lx * 256 + DIH + d];
    yg[Lx * DIH + d] = (_Float16)(y * silu_f(z));
  }
}

// ---------------- recombine: concat(of, ob) + residual, scatter back to volume ----------------
__global__ void k_recombine(const float* __restrict__ curIn,
                            const float* __restrict__ outf, const float* __restrict__ outb,
                            float* __restrict__ curOut, int mode) {
  int idx = blockIdx.x * blockDim.x + threadIdx.x;   // over LSEQ*CCH
  int l = idx >> 7, c = idx & 127;
  int off = spatial_off(mode, l);
  float m = (c < DMH) ? outf[l * DMH + c] : outb[l * DMH + (c - DMH)];
  curOut[c * LSEQ + off] = curIn[c * LSEQ + off] + m;
}

__global__ void k_final(const float* __restrict__ cur, const float* __restrict__ x,
                        float* __restrict__ out) {
  int i = blockIdx.x * blockDim.x + threadIdx.x;
  out[i] = cur[i] + x[i];
}

// ---------------- host launcher ----------------
extern "C" void kernel_launch(void* const* d_in, const int* in_sizes, int n_in,
                              void* d_out, int out_size, void* d_ws, size_t ws_size,
                              hipStream_t stream) {
  const float* x    = (const float*)d_in[0];
  const float* inw  = (const float*)d_in[1];   // (6,256,64)
  const float* cw   = (const float*)d_in[2];   // (6,128,4)
  const float* cb   = (const float*)d_in[3];   // (6,128)
  const float* xpw  = (const float*)d_in[4];   // (6,20,128)
  const float* dpw  = (const float*)d_in[5];   // (6,128,4)
  const float* dpb  = (const float*)d_in[6];   // (6,128)
  const float* alog = (const float*)d_in[7];   // (6,128,8)
  const float* dpar = (const float*)d_in[8];   // (6,128)
  const float* opw  = (const float*)d_in[9];   // (6,64,128)
  const float* lng  = (const float*)d_in[10];  // (3,128)
  const float* lnb  = (const float*)d_in[11];  // (3,128)

  float* ws   = (float*)d_ws;
  float* curA = ws;                              // LSEQ*128 f32
  float* curB = curA + LSEQ * CCH;               // LSEQ*128 f32
  float* xz   = curB + LSEQ * CCH;               // LSEQ*256 f32
  float* xc   = xz   + LSEQ * 256;               // LSEQ*128 f32
  float* dbc  = xc   + LSEQ * DIH;               // LSEQ*20  f32
  float* dt   = dbc  + LSEQ * 20;                // LSEQ*128 f32
  float* aggA = dt   + LSEQ * DIH;               // 216*1024 f32
  float* aggB = aggA + NCHUNK * 1024;            // 216*1024 f32
  float* outf = aggB + NCHUNK * 1024;            // LSEQ*64  f32
  float* outb = outf + LSEQ * DMH;               // LSEQ*64  f32
  _Float16* xa_h  = (_Float16*)(outb + LSEQ * DMH);   // LSEQ*64  f16
  _Float16* xbr_h = xa_h  + LSEQ * DMH;               // LSEQ*64  f16
  _Float16* yg_h  = xbr_h + LSEQ * DMH;               // LSEQ*128 f16
  _Float16* inw_h = yg_h  + LSEQ * DIH;               // 256*64   f16
  _Float16* opw_h = inw_h + 256 * 64;                 // 64*128   f16

  const float* curIn = x;
  float* pp[2] = {curA, curB};
  for (int mode = 0; mode < 3; ++mode) {
    float* curOut = pp[mode & 1];
    k_prep_ln<<<LSEQ, 128, 0, stream>>>(curIn, xa_h, xbr_h,
                                        lng + mode * CCH, lnb + mode * CCH, mode);
    for (int dir = 0; dir < 2; ++dir) {
      int wi = 2 * mode + dir;
      const _Float16* Ain = dir ? xbr_h : xa_h;
      float* Oout = dir ? outb : outf;
      k_cvt16<<<64, 256, 0, stream>>>(inw + wi * 256 * 64, inw_h, 256 * 64);
      k_cvt16<<<32, 256, 0, stream>>>(opw + wi * 64 * 128, opw_h, 64 * 128);
      // in_proj: (L,64) x (256,64)^T -> (L,256)   [NT=8, grid.y=2 covers N=256]
      k_gemm_h<64, 8><<<dim3(LSEQ / 64, 2), 128, 0, stream>>>(
          Ain, inw_h, xz, 256, LSEQ, 0);
      k_conv_silu<<<(LSEQ * DIH) / 256, 256, 0, stream>>>(
          xz, cw + wi * DIH * 4, cb + wi * DIH, xc);
      k_proj_dt<<<LSEQ, 128, 0, stream>>>(
          xc, xpw + wi * 20 * DIH, dpw + wi * DIH * DTRN, dpb + wi * DIH, dbc, dt);
      k_scan1<<<NCHUNK, 128, 0, stream>>>(dt, xc, dbc, alog + wi * DIH * DSN, aggA, aggB);
      k_scan2<<<4, 256, 0, stream>>>(aggA, aggB);
      k_scan3<<<NCHUNK, 128, 0, stream>>>(dt, xc, dbc, alog + wi * DIH * DSN,
                                          dpar + wi * DIH, aggB, xz, yg_h);
      // out_proj: (L,128) x (64,128)^T -> (L,64)  [NT=4 covers N=64]; reverse rows for dir=1
      k_gemm_h<128, 4><<<dim3(LSEQ / 64, 1), 128, 0, stream>>>(
          yg_h, opw_h, Oout, DMH, LSEQ, dir);
    }
    k_recombine<<<(LSEQ * CCH) / 256, 256, 0, stream>>>(curIn, outf, outb, curOut, mode);
    curIn = curOut;
  }
  k_final<<<(LSEQ * CCH) / 256, 256, 0, stream>>>(curIn, x, (float*)d_out);
}
